// MultiHeadAttentionBern_87720412053906
// MI455X (gfx1250) — compile-verified, matrix operand order checked by reading, not run on hardware
//
#include <hip/hip_runtime.h>
#include <hip/hip_bf16.h>
#include <stdint.h>

// ---- problem constants (from reference) ----
#define B_  4
#define S_  1024
#define E_  768
#define H_  12
#define DK_ 64

typedef __bf16 bf16_t;
typedef __attribute__((ext_vector_type(16))) __bf16 v16bf;
typedef __attribute__((ext_vector_type(8)))  float  v8f;
typedef __attribute__((ext_vector_type(4)))  int    i32x4;
typedef __attribute__((ext_vector_type(8)))  int    i32x8;

// ---------------- PRNG / gumbel ----------------
static __device__ __forceinline__ uint32_t pcg_hash(uint32_t x) {
    x = x * 747796405u + 2891336453u;
    uint32_t w = ((x >> ((x >> 28u) + 4u)) ^ x) * 277803737u;
    return (w >> 22u) ^ w;
}
static __device__ __forceinline__ float gumbel_from(uint32_t h) {
    float u = ((float)h + 0.5f) * 2.3283064365386963e-10f;  // (0,1)
    return -__logf(-__logf(u));
}

// ---------------- vectorized fragment-chunk loaders ----------------
// 8 consecutive bf16 (16 bytes) -> fragment elements [lo..lo+7]
static __device__ __forceinline__ void load8(v16bf& f, int lo, const bf16_t* p) {
    union { uint4 u; bf16_t h[8]; } c;
    c.u = *(const uint4*)p;               // global_load_b128 / ds_load_b128
#pragma unroll
    for (int j = 0; j < 8; ++j) f[lo + j] = c.h[j];
}
// 8 consecutive fp32 (two b128 loads) -> bf16 fragment elements [lo..lo+7]
static __device__ __forceinline__ void load8f(v16bf& f, int lo, const float* p) {
    float4 x = *(const float4*)p;
    float4 y = *(const float4*)(p + 4);
    f[lo + 0] = (bf16_t)x.x; f[lo + 1] = (bf16_t)x.y;
    f[lo + 2] = (bf16_t)x.z; f[lo + 3] = (bf16_t)x.w;
    f[lo + 4] = (bf16_t)y.x; f[lo + 5] = (bf16_t)y.y;
    f[lo + 6] = (bf16_t)y.z; f[lo + 7] = (bf16_t)y.w;
}

// ---------------- Tensor Data Mover (TDM) helpers ----------------
// Flat shared-memory addresses truncate to the LDS byte offset (ISA 10.2 aperture rules).
static __device__ __forceinline__ uint32_t lds_addr_of(const void* p) {
    return (uint32_t)(size_t)p;
}
// 2D bf16 tile load: rows x rowlen elements, row stride in elements.
// Builds D# group0 (count/lds/global/type) + group1 (sizes/dims/strides) in SGPRs
// and issues tensor_load_to_lds (tracked by TENSORcnt).
static __device__ __forceinline__ void tdm_load_2d_bf16(uint32_t lds_byte_addr,
                                                        const void* gsrc,
                                                        uint32_t rows,
                                                        uint32_t rowlen,
                                                        uint32_t row_stride,
                                                        uint32_t tensor_rows)
{
    uint64_t ga = (uint64_t)(size_t)gsrc;
    i32x4 g0;
    g0[0] = 1;                                    // count=1 (valid user descriptor)
    g0[1] = (int)lds_byte_addr;                   // lds_addr
    g0[2] = (int)(uint32_t)ga;                    // global_addr[31:0]
    g0[3] = (int)(((uint32_t)(ga >> 32) & 0x01FFFFFFu) | (2u << 30)); // addr[56:32] | type=2

    uint64_t stride0 = row_stride;                // dim0 stride (elements)
    i32x8 g1;
    g1[0] = (int)(1u << 16);                      // data_size=1 -> 2-byte elements
    g1[1] = (int)((rowlen & 0xFFFFu) << 16);      // tensor_dim0[15:0] (ab_addr=0)
    g1[2] = (int)((rowlen >> 16) | ((tensor_rows & 0xFFFFu) << 16));   // dim0 hi | dim1 lo
    g1[3] = (int)(((tensor_rows >> 16) & 0xFFFFu) | ((rowlen & 0xFFFFu) << 16)); // dim1 hi | tile_dim0
    g1[4] = (int)(rows & 0xFFFFu);                // tile_dim1 | tile_dim2=0
    g1[5] = (int)(uint32_t)stride0;               // tensor_dim0_stride[31:0]
    g1[6] = (int)((uint32_t)(stride0 >> 32) & 0xFFFFu); // stride0 hi | stride1 lo = 0
    g1[7] = 0;                                    // stride1 hi (unused for 2D tile)

    asm volatile("tensor_load_to_lds %0, %1" :: "s"(g0), "s"(g1) : "memory");
}
template <int N>
static __device__ __forceinline__ void tdm_wait() {
#if defined(__has_builtin)
#if __has_builtin(__builtin_amdgcn_s_wait_tensorcnt)
    __builtin_amdgcn_s_wait_tensorcnt((unsigned short)N);
#define TDM_WAIT_DONE_ 1
#endif
#endif
#ifndef TDM_WAIT_DONE_
    if (N == 0) asm volatile("s_wait_tensorcnt 0x0" ::: "memory");
    else        asm volatile("s_wait_tensorcnt 0x1" ::: "memory");
#endif
}
static __device__ __forceinline__ void ds_drain() {
    asm volatile("s_wait_dscnt 0x0" ::: "memory");  // LDS reads done before TDM overwrites
}

// ---------------- projection GEMM: Y = (X @ W^T + bias) * scale ----------------
// X fp32 [M,K], W fp32 [N,K]; one wave computes a 16x16 tile via v_wmma bf16.
__global__ __launch_bounds__(32)
void proj_gemm_f32(const float* __restrict__ X,
                   const float* __restrict__ W,
                   const float* __restrict__ bias,
                   bf16_t* __restrict__ Y,
                   float scale, int headMajor)
{
    const int K = E_;
    const int n0 = blockIdx.x * 16;
    const int m0 = blockIdx.y * 16;
    const int lane = threadIdx.x & 31;
    const int hf   = lane >> 4;       // wave half
    const int mr   = lane & 15;       // row (A) / col (B,C)
    const int klo  = hf * 8;          // K sub-offset per ISA 16-bit A layout

    const float* Arow = X + (size_t)(m0 + mr) * K;
    const float* Brow = W + (size_t)(n0 + mr) * K;

    v8f c = {};
    for (int kb = 0; kb < K; kb += 32) {
        v16bf a, b;
        load8f(a, 0, Arow + kb + klo);
        load8f(a, 8, Arow + kb + 16 + klo);
        load8f(b, 0, Brow + kb + klo);
        load8f(b, 8, Brow + kb + 16 + klo);
        if (kb + 32 < K) {
            __builtin_prefetch(Arow + kb + 32, 0, 1);   // global_prefetch_b8
            __builtin_prefetch(Brow + kb + 32, 0, 1);
        }
        c = __builtin_amdgcn_wmma_f32_16x16x32_bf16(false, a, false, b,
                                                    (short)0, c, false, false);
    }

#pragma unroll
    for (int r = 0; r < 8; ++r) {
        int row = r + hf * 8;
        int m = m0 + row, n = n0 + mr;
        float v = (c[r] + bias[n]) * scale;
        size_t idx;
        if (headMajor) {  // [B,H,S,DK]
            int bb = m / S_, ss = m % S_, hh = n / DK_, dd = n % DK_;
            idx = (((size_t)(bb * H_ + hh) * S_) + ss) * DK_ + dd;
        } else {
            idx = (size_t)m * E_ + n;
        }
        Y[idx] = (bf16_t)v;
    }
}

// ---------------- output projection: out = hidden(bf16) @ Wp^T + bp (fp32) ----------------
__global__ __launch_bounds__(32)
void proj_gemm_out(const bf16_t* __restrict__ X,
                   const float* __restrict__ W,
                   const float* __restrict__ bias,
                   float* __restrict__ Y)
{
    const int K = E_;
    const int n0 = blockIdx.x * 16;
    const int m0 = blockIdx.y * 16;
    const int lane = threadIdx.x & 31;
    const int hf   = lane >> 4;
    const int mr   = lane & 15;
    const int klo  = hf * 8;

    const bf16_t* Arow = X + (size_t)(m0 + mr) * K;
    const float*  Brow = W + (size_t)(n0 + mr) * K;

    v8f c = {};
    for (int kb = 0; kb < K; kb += 32) {
        v16bf a, b;
        load8(a, 0, Arow + kb + klo);
        load8(a, 8, Arow + kb + 16 + klo);
        load8f(b, 0, Brow + kb + klo);
        load8f(b, 8, Brow + kb + 16 + klo);
        if (kb + 32 < K) {
            __builtin_prefetch(Arow + kb + 32, 0, 1);
            __builtin_prefetch(Brow + kb + 32, 0, 1);
        }
        c = __builtin_amdgcn_wmma_f32_16x16x32_bf16(false, a, false, b,
                                                    (short)0, c, false, false);
    }
#pragma unroll
    for (int r = 0; r < 8; ++r) {
        int row = r + hf * 8;
        Y[(size_t)(m0 + row) * E_ + n0 + mr] = c[r] + bias[n0 + mr];
    }
}

// ---------------- attention per (b, h, 16-row query tile) ----------------
// K/V tiles staged into LDS by the Tensor Data Mover, double-buffered so the
// DMA of tile i+1 overlaps the WMMAs of tile i. Logit block lives in LDS.
// LDS: 64KB logits + 2x2KB K + 2x4KB V = 76KB  (<< 320KB/WGP).
__global__ __launch_bounds__(32)
void attn_kernel(const bf16_t* __restrict__ qws,  // [B,H,S,DK], pre-scaled 1/sqrt(dk)
                 const bf16_t* __restrict__ kws,
                 const bf16_t* __restrict__ vws,
                 bf16_t* __restrict__ hws,        // [B,S,E] merged hidden
                 float* __restrict__ maskOut,     // [B,S,S] (pre-zeroed)
                 float* __restrict__ adjOut)      // [B,S,S] (pre-zeroed)
{
    __shared__ float slog[16 * S_];
    __shared__ __align__(16) bf16_t stK[2][16 * DK_];
    __shared__ __align__(16) bf16_t stV[2][32 * DK_];

    const int b  = blockIdx.z;
    const int h  = blockIdx.y;
    const int s0 = blockIdx.x * 16;
    const int lane = threadIdx.x & 31;
    const int hf   = lane >> 4;
    const int mr   = lane & 15;
    const int klo  = hf * 8;

    const size_t headBase = (size_t)(b * H_ + h) * S_ * DK_;
    const bf16_t* kHead = kws + headBase;
    const bf16_t* vHead = vws + headBase;

    // q fragments covering d = 0..31 and d = 32..63 (b128 loads)
    const bf16_t* qrow = qws + headBase + (size_t)(s0 + mr) * DK_;
    v16bf aq0, aq1;
    load8(aq0, 0, qrow + klo);      load8(aq0, 8, qrow + 16 + klo);
    load8(aq1, 0, qrow + 32 + klo); load8(aq1, 8, qrow + 48 + klo);

    // ---- logits = q @ k^T : TDM-staged 16x64 K tiles, double buffered ----
    tdm_load_2d_bf16(lds_addr_of(stK[0]), kHead, 16, DK_, DK_, S_);
    for (int tt = 0; tt < S_ / 16; ++tt) {
        const int cur = tt & 1;
        if (tt + 1 < S_ / 16) {
            ds_drain();
            tdm_load_2d_bf16(lds_addr_of(stK[(tt + 1) & 1]),
                             kHead + (size_t)(tt + 1) * 16 * DK_, 16, DK_, DK_, S_);
            tdm_wait<1>();          // oldest (current) tile complete
        } else {
            tdm_wait<0>();
        }
        const bf16_t* krow = stK[cur] + mr * DK_;
        v16bf bk0, bk1;
        load8(bk0, 0, krow + klo);      load8(bk0, 8, krow + 16 + klo);
        load8(bk1, 0, krow + 32 + klo); load8(bk1, 8, krow + 48 + klo);

        v8f c = {};
        c = __builtin_amdgcn_wmma_f32_16x16x32_bf16(false, aq0, false, bk0,
                                                    (short)0, c, false, false);
        c = __builtin_amdgcn_wmma_f32_16x16x32_bf16(false, aq1, false, bk1,
                                                    (short)0, c, false, false);
#pragma unroll
        for (int r = 0; r < 8; ++r)
            slog[(r + hf * 8) * S_ + tt * 16 + mr] = c[r];
    }
    __syncthreads();

    // ---- row softmax stats: lane r (<16) owns row r ----
    float rmax = -3.4e38f, rsum = 0.f;
    if (lane < 16) {
        const float* row = &slog[lane * S_];
        for (int t = 0; t < S_; ++t) rmax = fmaxf(rmax, row[t]);
        for (int t = 0; t < S_; ++t) rsum += __expf(row[t] - rmax);
    }
    const float myMax = __shfl(rmax, mr, 32);
    const float myInv = 1.0f / __shfl(rsum, mr, 32);

    // ---- hidden = (A * probs) @ V ; TDM-staged 32x64 V tiles, double buffered ----
    v8f zero = {};
    v8f ch[4] = { zero, zero, zero, zero };
    const uint32_t rowSeedBase = ((uint32_t)(b * H_ + h) << 20) ^ ((uint32_t)(s0 + mr) << 10);

    tdm_load_2d_bf16(lds_addr_of(stV[0]), vHead, 32, DK_, DK_, S_);
    for (int tb = 0; tb < S_; tb += 32) {
        const int cur = (tb >> 5) & 1;
        if (tb + 32 < S_) {
            ds_drain();
            tdm_load_2d_bf16(lds_addr_of(stV[((tb >> 5) + 1) & 1]),
                             vHead + (size_t)(tb + 32) * DK_, 32, DK_, DK_, S_);
            tdm_wait<1>();
        } else {
            tdm_wait<0>();
        }

        v16bf am;
#pragma unroll
        for (int j = 0; j < 16; ++j) {
            int toff = (j < 8) ? (klo + j) : (16 + klo + (j - 8));
            int t = tb + toff;
            float lg   = slog[mr * S_ + t];
            float prob = __expf(lg - myMax) * myInv;
            uint32_t seed = rowSeedBase ^ (uint32_t)t;
            float g1 = gumbel_from(pcg_hash(seed ^ 0x9E3779B9u));
            float g0 = gumbel_from(pcg_hash(seed ^ 0x7F4A7C15u));
            float d  = lg + g1 - g0;              // TAU = 1
            float Av = (d > 0.f) ? 1.f : 0.f;     // hard ST forward value
            am[j] = (bf16_t)(Av * prob);
        }
#pragma unroll
        for (int nt = 0; nt < 4; ++nt) {
            v16bf bv;
#pragma unroll
            for (int j = 0; j < 16; ++j) {
                int toff = (j < 8) ? (klo + j) : (16 + klo + (j - 8));
                bv[j] = stV[cur][toff * DK_ + nt * 16 + mr];
            }
            ch[nt] = __builtin_amdgcn_wmma_f32_16x16x32_bf16(false, am, false, bv,
                                                             (short)0, ch[nt], false, false);
        }
    }

    // store merged hidden [B,S,E] as bf16 for the output projection
#pragma unroll
    for (int nt = 0; nt < 4; ++nt) {
#pragma unroll
        for (int r = 0; r < 8; ++r) {
            int row = r + hf * 8;
            size_t idx = ((size_t)b * S_ + s0 + row) * E_ + h * DK_ + nt * 16 + mr;
            hws[idx] = (bf16_t)ch[nt][r];
        }
    }

    // ---- head-summed mask / adjacency (atomic accumulate over h) ----
    for (int m = 0; m < 16; ++m) {
        float rM = __shfl(rmax, m, 32);
        float rI = 1.0f / __shfl(rsum, m, 32);
        int s = s0 + m;
        uint32_t sb = ((uint32_t)(b * H_ + h) << 20) ^ ((uint32_t)s << 10);
        for (int t = lane; t < S_; t += 32) {
            float lg   = slog[m * S_ + t];
            float prob = __expf(lg - rM) * rI;
            uint32_t seed = sb ^ (uint32_t)t;
            float g1 = gumbel_from(pcg_hash(seed ^ 0x9E3779B9u));
            float g0 = gumbel_from(pcg_hash(seed ^ 0x7F4A7C15u));
            float d  = lg + g1 - g0;
            float Av = (d > 0.f) ? 1.f : 0.f;
            size_t o = ((size_t)b * S_ + s) * S_ + t;
            atomicAdd(&maskOut[o], Av);
            atomicAdd(&adjOut[o], Av * prob);
        }
    }
}

// ---------------- utility ----------------
__global__ void zero_kernel(float* __restrict__ p, size_t n) {
    size_t i = (size_t)blockIdx.x * blockDim.x + threadIdx.x;
    if (i < n) p[i] = 0.f;
}

// ---------------- host side ----------------
extern "C" void kernel_launch(void* const* d_in, const int* in_sizes, int n_in,
                              void* d_out, int out_size, void* d_ws, size_t ws_size,
                              hipStream_t stream)
{
    const float* queries = (const float*)d_in[0];
    const float* keys    = (const float*)d_in[1];
    const float* values  = (const float*)d_in[2];
    const float* Wq = (const float*)d_in[3];
    const float* bq = (const float*)d_in[4];
    const float* Wk = (const float*)d_in[5];
    const float* bk = (const float*)d_in[6];
    const float* Wv = (const float*)d_in[7];
    const float* bv = (const float*)d_in[8];
    const float* Wp = (const float*)d_in[9];
    const float* bp = (const float*)d_in[10];

    float* out     = (float*)d_out;                       // [B,S,E]
    float* maskOut = out + (size_t)B_ * S_ * E_;          // [B,S,S]
    float* adjOut  = maskOut + (size_t)B_ * S_ * S_;      // [B,S,S]

    const size_t perTensor = (size_t)B_ * H_ * S_ * DK_;  // == B*S*E
    bf16_t* qws = (bf16_t*)d_ws;
    bf16_t* kws = qws + perTensor;
    bf16_t* vws = kws + perTensor;
    bf16_t* hws = vws + perTensor;

    const dim3 gproj(E_ / 16, (B_ * S_) / 16);            // one wave per 16x16 tile

    proj_gemm_f32<<<gproj, 32, 0, stream>>>(queries, Wq, bq, qws, 0.125f, 1); // 1/sqrt(64)
    proj_gemm_f32<<<gproj, 32, 0, stream>>>(keys,    Wk, bk, kws, 1.0f,   1);
    proj_gemm_f32<<<gproj, 32, 0, stream>>>(values,  Wv, bv, vws, 1.0f,   1);

    const size_t nz = (size_t)2 * B_ * S_ * S_;           // mask + adjacency
    zero_kernel<<<(unsigned)((nz + 255) / 256), 256, 0, stream>>>(maskOut, nz);

    const dim3 gattn(S_ / 16, H_, B_);
    attn_kernel<<<gattn, 32, 0, stream>>>(qws, kws, vws, hws, maskOut, adjOut);

    proj_gemm_out<<<gproj, 32, 0, stream>>>(hws, Wp, bp, out);
}
